// VanillaRNN_89610197664195
// MI455X (gfx1250) — matrix operand order repaired and checked by reading, not compiled, not run on hardware
//
#include <hip/hip_runtime.h>
#include <stdint.h>
#include <stddef.h>

// ---------------------------------------------------------------------------
// VanillaRNN for MI455X (gfx1250): bf16 WMMA (f32 accum), 512 sequential step
// kernels + weight-convert prologue + final projection. wave32 everywhere.
// Round 3: K-chunk = 64 (8 barriers/step instead of 16; 8 WMMAs per barrier),
// double-buffered LDS staging, branch-free tanh, +64-row x_t prefetch.
// ---------------------------------------------------------------------------

typedef __attribute__((ext_vector_type(16))) __bf16 v16bf;
typedef __attribute__((ext_vector_type(8)))  float  v8f;

#define HID  512
#define SEQ  512
#define NCLS 1000

// fp32 -> bf16, round-to-nearest-even.
__device__ __forceinline__ uint16_t f2bf(float f) {
  uint32_t x = __builtin_bit_cast(uint32_t, f);
  x += 0x7fffu + ((x >> 16) & 1u);
  return (uint16_t)(x >> 16);
}

// Branch-free tanh: tanh(v) = 1 - 2/(exp2(v * 2*log2(e)) + 1).
// exp2->0 gives -1, exp2->inf gives +1; ~1e-7 rel error, far below bf16 ulp.
__device__ __forceinline__ float fast_tanh(float v) {
  const float e = __builtin_amdgcn_exp2f(v * 2.8853900817779268f);
  return 1.0f - 2.0f * __builtin_amdgcn_rcpf(e + 1.0f);
}

__device__ __forceinline__ v16bf frag_cast(uint4 lo, uint4 hi) {
  union { uint4 q[2]; v16bf v; } u;
  u.q[0] = lo; u.q[1] = hi;
  return u.v;
}

// Store 4 packed (K even | K odd<<16) dwords for columns ncol..ncol+3 into a
// 32K x 64N LDS sub-panel held in the exact WMMA B-operand per-lane layout:
//   tile nt (16 cols) -> 1KB; lane = ((k>>3)&1)*16 + (n&15); dword v = ((k&7)>>1) + 4*(k>>4).
__device__ __forceinline__ void stage_store(uint32_t* lds, int ncol, int half_st,
                                            int vg, uint32_t p0, uint32_t p1,
                                            uint32_t p2, uint32_t p3) {
  const int nt = ncol >> 4;
  const int lb = half_st * 16 + (ncol & 15);
  uint32_t* d = lds + nt * 256 + vg;
  d[(lb + 0) * 8] = p0;
  d[(lb + 1) * 8] = p1;
  d[(lb + 2) * 8] = p2;
  d[(lb + 3) * 8] = p3;
}

// Stage a 32K x 64N sub-panel of x_t (fp32, rows 1MB apart) -> bf16 frag layout.
__device__ __forceinline__ void stage_x_panel(uint32_t* lds, const float* xt,
                                              int kc, int k0, int nblock,
                                              int ncol, int half_st, int vg) {
  const float* r0 = xt + (size_t)(kc + k0) * (SEQ * HID) + nblock + ncol;
  const float4 a = *(const float4*)r0;
  const float4 b = *(const float4*)(r0 + SEQ * HID);
  if (kc + 64 < HID)  // pull next round's far-stride rows toward L2/WGP$
    __builtin_prefetch(r0 + 64 * (size_t)(SEQ * HID), 0, 1);
  stage_store(lds, ncol, half_st, vg,
              (uint32_t)f2bf(a.x) | ((uint32_t)f2bf(b.x) << 16),
              (uint32_t)f2bf(a.y) | ((uint32_t)f2bf(b.y) << 16),
              (uint32_t)f2bf(a.z) | ((uint32_t)f2bf(b.z) << 16),
              (uint32_t)f2bf(a.w) | ((uint32_t)f2bf(b.w) << 16));
}

// Stage a 32K x 64N sub-panel of a bf16 row-major matrix -> frag layout.
__device__ __forceinline__ void stage_h_panel(uint32_t* lds, const uint16_t* H,
                                              int kc, int k0, int nblock,
                                              int ncol, int half_st, int vg) {
  const uint16_t* r0 = H + (size_t)(kc + k0) * HID + nblock + ncol;
  const uint2 a = *(const uint2*)r0;
  const uint2 b = *(const uint2*)(r0 + HID);
  stage_store(lds, ncol, half_st, vg,
              (a.x & 0xffffu) | (b.x << 16),
              (a.x >> 16)     | (b.x & 0xffff0000u),
              (a.y & 0xffffu) | (b.y << 16),
              (a.y >> 16)     | (b.y & 0xffff0000u));
}

// ---------------------------------------------------------------------------
// Prologue: convert weights to bf16 in workspace, zero h0.
// ---------------------------------------------------------------------------
__global__ __launch_bounds__(256) void cvt_weights(
    const float* __restrict__ Whx, const float* __restrict__ Whh,
    const float* __restrict__ Wph,
    uint16_t* __restrict__ whx, uint16_t* __restrict__ whh,
    uint16_t* __restrict__ wph, uint16_t* __restrict__ h0) {
  const int idx = blockIdx.x * 256 + threadIdx.x;
  if (idx < HID * HID) {
    whx[idx] = f2bf(Whx[idx]);
    whh[idx] = f2bf(Whh[idx]);
    h0[idx]  = 0;  // h starts at zero every launch (deterministic)
  }
  if (idx < NCLS * HID) wph[idx] = f2bf(Wph[idx]);
}

// ---------------------------------------------------------------------------
// One RNN step: Hout = tanh(Whx @ x_t + Whh @ Hin + bh)   (all 512x512, K=512)
// Block = 256 threads (8 waves), 64x64 output tile, wave = 16(M) x 32(N).
// Double-buffered 64-K LDS panels: 8 barriers/step, 8 WMMAs per barrier,
// staging of the next panel overlaps the current panel's WMMAs.
// ---------------------------------------------------------------------------
__global__ __launch_bounds__(256) void rnn_step(
    const uint16_t* __restrict__ Whx, const uint16_t* __restrict__ Whh,
    const float* __restrict__ x, const float* __restrict__ bh,
    const uint16_t* __restrict__ Hin, uint16_t* __restrict__ Hout, int t) {
  __shared__ uint32_t ldsB1[4096];  // x_t panel: 2 buffers x (2 sub-chunks x 4KB)
  __shared__ uint32_t ldsB2[4096];  // H   panel: 2 buffers x (2 sub-chunks x 4KB)

  const int tid  = threadIdx.x;
  const int lane = tid & 31;
  const int wave = tid >> 5;
  const int mt = wave & 3;   // 16-row M tile within the 64-row block tile
  const int np = wave >> 2;  // which pair of 16-col N tiles
  const int mbase  = blockIdx.y * 64 + mt * 16;
  const int nblock = blockIdx.x * 64;
  const int l15 = lane & 15;
  const int lh  = lane >> 4;

  // staging decomposition: 256 threads = 16 K-pairs x 16 groups of 4 columns
  const int n4      = tid & 15;
  const int kp      = tid >> 4;
  const int k0      = kp << 1;
  const int half_st = (kp >> 2) & 1;
  const int vg      = (kp & 3) | ((kp >> 3) << 2);
  const int ncol    = n4 << 2;

  v8f acc0 = {0.f, 0.f, 0.f, 0.f, 0.f, 0.f, 0.f, 0.f};
  v8f acc1 = acc0;

  // x is (hid, seq, in): x_t[k][n] = x[k*SEQ*HID + t*HID + n]
  const float* xt = x + (size_t)t * HID;

  // prime buffer 0 with K = 0..63
  stage_x_panel(ldsB1,        xt, 0,  k0, nblock, ncol, half_st, vg);
  stage_x_panel(ldsB1 + 1024, xt, 32, k0, nblock, ncol, half_st, vg);
  stage_h_panel(ldsB2,        Hin, 0,  k0, nblock, ncol, half_st, vg);
  stage_h_panel(ldsB2 + 1024, Hin, 32, k0, nblock, ncol, half_st, vg);

  for (int kc = 0; kc < HID; kc += 64) {
    const int cur = (kc >> 6) & 1;
    uint32_t* rB1 = ldsB1 + cur * 2048;
    uint32_t* rB2 = ldsB2 + cur * 2048;
    __syncthreads();  // staged data for `cur` visible; prior reads of cur^1 done

    if (kc + 64 < HID) {  // stage next 64-K panel into the other buffer
      uint32_t* w1 = ldsB1 + (cur ^ 1) * 2048;
      uint32_t* w2 = ldsB2 + (cur ^ 1) * 2048;
      stage_x_panel(w1,        xt, kc + 64, k0, nblock, ncol, half_st, vg);
      stage_x_panel(w1 + 1024, xt, kc + 96, k0, nblock, ncol, half_st, vg);
      stage_h_panel(w2,        Hin, kc + 64, k0, nblock, ncol, half_st, vg);
      stage_h_panel(w2 + 1024, Hin, kc + 96, k0, nblock, ncol, half_st, vg);
    }

#pragma unroll
    for (int sub = 0; sub < 2; ++sub) {
      const int kk = kc + sub * 32;
      // A fragments straight from global (row-major bf16, ld = 512):
      // lane l: row = l&15; VGPRs 0-3 = K[kk + 8*(l>>4)..], VGPRs 4-7 = +16.
      const uint16_t* pa1 = Whx + (size_t)(mbase + l15) * HID + kk + lh * 8;
      const uint16_t* pa2 = Whh + (size_t)(mbase + l15) * HID + kk + lh * 8;
      const v16bf A1 = frag_cast(*(const uint4*)pa1, *(const uint4*)(pa1 + 16));
      const v16bf A2 = frag_cast(*(const uint4*)pa2, *(const uint4*)(pa2 + 16));
      const uint32_t* sB1 = rB1 + sub * 1024;
      const uint32_t* sB2 = rB2 + sub * 1024;

#pragma unroll
      for (int j = 0; j < 2; ++j) {
        const int nt = np * 2 + j;
        const uint4* b1 = (const uint4*)sB1 + nt * 64 + lane * 2;
        const uint4* b2 = (const uint4*)sB2 + nt * 64 + lane * 2;
        const v16bf B1 = frag_cast(b1[0], b1[1]);
        const v16bf B2 = frag_cast(b2[0], b2[1]);
        v8f acc = j ? acc1 : acc0;
        acc = __builtin_amdgcn_wmma_f32_16x16x32_bf16(false, A1, false, B1,
                                                      (short)0, acc, false, false);
        acc = __builtin_amdgcn_wmma_f32_16x16x32_bf16(false, A2, false, B2,
                                                      (short)0, acc, false, false);
        if (j) acc1 = acc; else acc0 = acc;
      }
    }
  }

  // epilogue: + bh, tanh, bf16 store. C/D layout: n = lane&15, m = 8*(l>>4)+g.
  const int mrow = mbase + lh * 8;
#pragma unroll
  for (int j = 0; j < 2; ++j) {
    const v8f acc = j ? acc1 : acc0;
    const int n = nblock + (np * 2 + j) * 16 + l15;
#pragma unroll
    for (int g = 0; g < 8; ++g) {
      const float v = fast_tanh(acc[g] + bh[mrow + g]);
      Hout[(size_t)(mrow + g) * HID + n] = f2bf(v);
    }
  }
}

// ---------------------------------------------------------------------------
// Final projection: out[b][c] = sum_k Wph[c][k] * H[k][b] + bp[c]
// M = classes (1000, guarded), N = batch (512), K = 512.
// ---------------------------------------------------------------------------
__global__ __launch_bounds__(256) void final_proj(
    const uint16_t* __restrict__ Wph, const uint16_t* __restrict__ H,
    const float* __restrict__ bp, float* __restrict__ out) {
  __shared__ uint32_t ldsB[4096];  // H panel: 2 buffers x (2 sub-chunks x 4KB)

  const int tid  = threadIdx.x;
  const int lane = tid & 31;
  const int wave = tid >> 5;
  const int mt = wave & 3;
  const int np = wave >> 2;
  const int mbase  = blockIdx.y * 64 + mt * 16;  // class tile base (may >= 1000)
  const int nblock = blockIdx.x * 64;            // batch tile base
  const int l15 = lane & 15;
  const int lh  = lane >> 4;

  const int n4      = tid & 15;
  const int kp      = tid >> 4;
  const int k0      = kp << 1;
  const int half_st = (kp >> 2) & 1;
  const int vg      = (kp & 3) | ((kp >> 3) << 2);
  const int ncol    = n4 << 2;

  v8f acc0 = {0.f, 0.f, 0.f, 0.f, 0.f, 0.f, 0.f, 0.f};
  v8f acc1 = acc0;

  const bool arow_ok = (mbase + l15) < NCLS;

  stage_h_panel(ldsB,        H, 0,  k0, nblock, ncol, half_st, vg);
  stage_h_panel(ldsB + 1024, H, 32, k0, nblock, ncol, half_st, vg);

  for (int kc = 0; kc < HID; kc += 64) {
    const int cur = (kc >> 6) & 1;
    uint32_t* rB = ldsB + cur * 2048;
    __syncthreads();

    if (kc + 64 < HID) {
      uint32_t* w = ldsB + (cur ^ 1) * 2048;
      stage_h_panel(w,        H, kc + 64, k0, nblock, ncol, half_st, vg);
      stage_h_panel(w + 1024, H, kc + 96, k0, nblock, ncol, half_st, vg);
    }

#pragma unroll
    for (int sub = 0; sub < 2; ++sub) {
      const int kk = kc + sub * 32;
      uint4 alo = {0u, 0u, 0u, 0u}, ahi = {0u, 0u, 0u, 0u};
      if (arow_ok) {
        const uint16_t* pa = Wph + (size_t)(mbase + l15) * HID + kk + lh * 8;
        alo = *(const uint4*)pa;
        ahi = *(const uint4*)(pa + 16);
      }
      const v16bf A = frag_cast(alo, ahi);
      const uint32_t* sB = rB + sub * 1024;

#pragma unroll
      for (int j = 0; j < 2; ++j) {
        const int nt = np * 2 + j;
        const uint4* bptr = (const uint4*)sB + nt * 64 + lane * 2;
        const v16bf B = frag_cast(bptr[0], bptr[1]);
        v8f acc = j ? acc1 : acc0;
        acc = __builtin_amdgcn_wmma_f32_16x16x32_bf16(false, A, false, B,
                                                      (short)0, acc, false, false);
        if (j) acc1 = acc; else acc0 = acc;
      }
    }
  }

  // out is (batch, classes) fp32; per lane the 8 accum values are contiguous
  // classes c = mbase + 8*lh + g at fixed batch row b.
  const int cbase = mbase + lh * 8;
#pragma unroll
  for (int j = 0; j < 2; ++j) {
    const v8f acc = j ? acc1 : acc0;
    const int b = nblock + (np * 2 + j) * 16 + l15;
#pragma unroll
    for (int g = 0; g < 8; ++g) {
      const int c = cbase + g;
      if (c < NCLS) out[(size_t)b * NCLS + c] = acc[g] + bp[c];
    }
  }
}

// ---------------------------------------------------------------------------
// Host-side launch: 1 convert + 512 steps (ping-pong h in ws) + 1 projection.
// ---------------------------------------------------------------------------
extern "C" void kernel_launch(void* const* d_in, const int* in_sizes, int n_in,
                              void* d_out, int out_size, void* d_ws, size_t ws_size,
                              hipStream_t stream) {
  const float* x   = (const float*)d_in[0];
  const float* Whx = (const float*)d_in[1];
  const float* Whh = (const float*)d_in[2];
  const float* bh  = (const float*)d_in[3];
  const float* Wph = (const float*)d_in[4];
  const float* bp  = (const float*)d_in[5];
  float* out = (float*)d_out;

  // workspace layout (bytes): whx_b[512K] whh_b[512K] wph_b[1M] h0[512K] h1[512K]
  char* ws = (char*)d_ws;
  uint16_t* whx_b = (uint16_t*)(ws);
  uint16_t* whh_b = (uint16_t*)(ws + 512 * 1024);
  uint16_t* wph_b = (uint16_t*)(ws + 1024 * 1024);
  uint16_t* h0    = (uint16_t*)(ws + 2048 * 1024);
  uint16_t* h1    = (uint16_t*)(ws + 2560 * 1024);

  cvt_weights<<<2000, 256, 0, stream>>>(Whx, Whh, Wph, whx_b, whh_b, wph_b, h0);

  uint16_t* bufs[2] = {h0, h1};
  for (int t = 0; t < SEQ; ++t) {
    rnn_step<<<dim3(8, 8), 256, 0, stream>>>(whx_b, whh_b, x, bh,
                                             bufs[t & 1], bufs[(t + 1) & 1], t);
  }
  // after 512 steps, final h lives in bufs[0]
  final_proj<<<dim3(8, 16), 256, 0, stream>>>(wph_b, bufs[0], bp, out);
}
